// Head_32719060861046
// MI455X (gfx1250) — matrix operand order repaired and checked by reading, not compiled, not run on hardware
//
#include <hip/hip_runtime.h>

#define Bb 8
#define Tt 2048
#define Dd 512
#define Hh 64

typedef __attribute__((ext_vector_type(16))) _Float16 v16h;
typedef __attribute__((ext_vector_type(8)))  _Float16 h8;
typedef __attribute__((ext_vector_type(8)))  float    v8f;
typedef __attribute__((ext_vector_type(4)))  float    f4;

static __device__ __forceinline__ v16h mk16(h8 lo, h8 hi) {
  v16h r;
#pragma unroll
  for (int i = 0; i < 8; ++i) { r[i] = lo[i]; r[8 + i] = hi[i]; }
  return r;
}
static __device__ __forceinline__ h8 ldh8(const _Float16* p) { return *(const h8*)p; }

// ---------------------------------------------------------------------------
// Kernel 1: transpose + convert weights to f16 [H, D] so WMMA B-operand rows
// are contiguous per lane.
// ---------------------------------------------------------------------------
__global__ void wprep_k(const float* __restrict__ Wk, const float* __restrict__ Wq,
                        const float* __restrict__ Wv, _Float16* __restrict__ WkT,
                        _Float16* __restrict__ WqT, _Float16* __restrict__ WvT) {
  int i = blockIdx.x * blockDim.x + threadIdx.x;
  if (i >= Hh * Dd) return;
  int h = i / Dd, d = i - h * Dd;
  WkT[i] = (_Float16)Wk[d * Hh + h];
  WqT[i] = (_Float16)Wq[d * Hh + h];
  WvT[i] = (_Float16)Wv[d * Hh + h];
}

// ---------------------------------------------------------------------------
// Kernel 2: QKV projection. One wave computes a 16-row tile for all three
// outputs (12 WMMA accumulators). x is the only DRAM-streamed tensor
// (33.5 MB) -> prefetch next K-chunk. q pre-scaled by H^-0.5; v stored
// transposed [B,H,T] with packed b128 stores.
// ---------------------------------------------------------------------------
__global__ __launch_bounds__(256) void qkvproj_k(
    const float* __restrict__ x,
    const _Float16* __restrict__ WqT, const _Float16* __restrict__ WkT,
    const _Float16* __restrict__ WvT,
    const float* __restrict__ bq, const float* __restrict__ bk,
    const float* __restrict__ bv,
    _Float16* __restrict__ qf, _Float16* __restrict__ kf,
    _Float16* __restrict__ vt) {
  const int lane = threadIdx.x & 31;
  const int wv   = threadIdx.x >> 5;
  const int wave = blockIdx.x * 8 + wv;
  const int hf = lane >> 4, ln = lane & 15;
  const int b  = wave >> 7;            // 128 row-tiles per batch
  const int t0 = (wave & 127) << 4;

  const float* xrow = x + (size_t)(b * Tt + t0 + ln) * Dd;
  const _Float16* Wt[3] = {WqT, WkT, WvT};

  v8f zero = {};
  v8f acc[3][4];
#pragma unroll
  for (int m = 0; m < 3; ++m)
#pragma unroll
    for (int n = 0; n < 4; ++n) acc[m][n] = zero;

  for (int kk = 0; kk < Dd; kk += 32) {
    // Prefetch next K-chunk of x (DRAM-streaming tensor).
    if (kk + 32 < Dd) __builtin_prefetch(xrow + kk + 32 + hf * 8, 0, 1);

    // A tile (x rows, f32 -> f16), ISA A-layout: lane<16 gets K {0..7,16..23}
    const int o = kk + hf * 8;
    f4 a0 = *(const f4*)(xrow + o);
    f4 a1 = *(const f4*)(xrow + o + 4);
    f4 a2 = *(const f4*)(xrow + o + 16);
    f4 a3 = *(const f4*)(xrow + o + 20);
    v16h A;
#pragma unroll
    for (int i = 0; i < 4; ++i) {
      A[i]      = (_Float16)a0[i];
      A[4 + i]  = (_Float16)a1[i];
      A[8 + i]  = (_Float16)a2[i];
      A[12 + i] = (_Float16)a3[i];
    }
#pragma unroll
    for (int m = 0; m < 3; ++m) {
#pragma unroll
      for (int nt = 0; nt < 4; ++nt) {
        const _Float16* wp = Wt[m] + (size_t)(nt * 16 + ln) * Dd + kk + hf * 8;
        v16h Bv = mk16(ldh8(wp), ldh8(wp + 16));
        acc[m][nt] = __builtin_amdgcn_wmma_f32_16x16x32_f16(
            false, A, false, Bv, (short)0, acc[m][nt], false, false);
      }
    }
  }

  // q: row-major f16, pre-scaled by 1/sqrt(H)
#pragma unroll
  for (int nt = 0; nt < 4; ++nt) {
    float bqv = bq[nt * 16 + ln];
#pragma unroll
    for (int r = 0; r < 8; ++r) {
      int row = t0 + r + hf * 8;
      qf[(size_t)(b * Tt + row) * Hh + nt * 16 + ln] =
          (_Float16)((acc[0][nt][r] + bqv) * 0.125f);
    }
  }
  // k: row-major f16
#pragma unroll
  for (int nt = 0; nt < 4; ++nt) {
    float bkv = bk[nt * 16 + ln];
#pragma unroll
    for (int r = 0; r < 8; ++r) {
      int row = t0 + r + hf * 8;
      kf[(size_t)(b * Tt + row) * Hh + nt * 16 + ln] =
          (_Float16)(acc[1][nt][r] + bkv);
    }
  }
  // v: transposed [B,H,T], 8 consecutive t per lane -> one b128 store per tile
#pragma unroll
  for (int nt = 0; nt < 4; ++nt) {
    float bvv = bv[nt * 16 + ln];
    h8 pk;
#pragma unroll
    for (int r = 0; r < 8; ++r) pk[r] = (_Float16)(acc[2][nt][r] + bvv);
    *(h8*)(vt + (size_t)(b * Hh + nt * 16 + ln) * Tt + t0 + hf * 8) = pk;
  }
}

// ---------------------------------------------------------------------------
// Kernel 3: flash attention. One independent wave per 16-row q tile,
// 32-column s-blocks, online softmax, P via per-wave LDS transpose.
// V tiles are loaded BEFORE the softmax so their latency hides under the
// shuffle/exp VALU work; next block's k/v lines are prefetched.
// ---------------------------------------------------------------------------
__global__ __launch_bounds__(256) void attn_k(
    const _Float16* __restrict__ qf, const _Float16* __restrict__ kf,
    const _Float16* __restrict__ vt, float* __restrict__ out) {
  __shared__ __align__(16) _Float16 Pbuf[8][16 * 32];  // 1KB per wave
  const int lane = threadIdx.x & 31;
  const int wv   = threadIdx.x >> 5;
  const int wave = blockIdx.x * 8 + wv;
  const int hf = lane >> 4, ln = lane & 15;
  const int b  = wave >> 7;
  const int i0 = (wave & 127) << 4;

  // q tile as A operand (K = H = 64 -> two K32 chunks), resident all loop
  const _Float16* qp = qf + (size_t)(b * Tt + i0 + ln) * Hh + hf * 16;
  v16h aq0 = mk16(ldh8(qp),      ldh8(qp + 8));
  v16h aq1 = mk16(ldh8(qp + 32), ldh8(qp + 40));

  v8f zero = {};
  v8f O[4];
#pragma unroll
  for (int nt = 0; nt < 4; ++nt) O[nt] = zero;
  float mrow[8], lrow[8];
#pragma unroll
  for (int r = 0; r < 8; ++r) { mrow[r] = -__builtin_inff(); lrow[r] = 0.f; }

  _Float16* P = &Pbuf[wv][0];
  const int send = i0 + 16;
  for (int s0 = 0; s0 < send; s0 += 32) {
    // k rows as B operand: lane = S-column, contiguous 16 halfs per K chunk
    const _Float16* kp0 = kf + (size_t)(b * Tt + s0 + ln) * Hh + hf * 16;
    const _Float16* kp1 = kp0 + 16 * Hh;
    v16h bk00 = mk16(ldh8(kp0),      ldh8(kp0 + 8));
    v16h bk01 = mk16(ldh8(kp0 + 32), ldh8(kp0 + 40));
    v16h bk10 = mk16(ldh8(kp1),      ldh8(kp1 + 8));
    v16h bk11 = mk16(ldh8(kp1 + 32), ldh8(kp1 + 40));

    // V tiles: independent of S -> issue now, consume after softmax
    v16h bV[4];
#pragma unroll
    for (int nt = 0; nt < 4; ++nt) {
      const _Float16* vp = vt + (size_t)(b * Hh + nt * 16 + ln) * Tt + s0 + hf * 16;
      bV[nt] = mk16(ldh8(vp), ldh8(vp + 8));
    }

    // Prefetch next s-block's k rows and v columns
    if (s0 + 32 < send) {
      __builtin_prefetch(kp0 + 32 * Hh, 0, 1);
      __builtin_prefetch(vt + (size_t)(b * Hh + ln) * Tt + s0 + 32 + hf * 16, 0, 1);
    }

    v8f S0 = zero, S1 = zero;
    S0 = __builtin_amdgcn_wmma_f32_16x16x32_f16(false, aq0, false, bk00, (short)0, S0, false, false);
    S0 = __builtin_amdgcn_wmma_f32_16x16x32_f16(false, aq1, false, bk01, (short)0, S0, false, false);
    S1 = __builtin_amdgcn_wmma_f32_16x16x32_f16(false, aq0, false, bk10, (short)0, S1, false, false);
    S1 = __builtin_amdgcn_wmma_f32_16x16x32_f16(false, aq1, false, bk11, (short)0, S1, false, false);

    if (s0 + 31 > i0) {  // diagonal block: causal mask (uniform branch)
      int c0 = s0 + ln, c1 = s0 + 16 + ln, rg = i0 + hf * 8;
#pragma unroll
      for (int r = 0; r < 8; ++r) {
        if (c0 > rg + r) S0[r] = -__builtin_inff();
        if (c1 > rg + r) S1[r] = -__builtin_inff();
      }
    }

    // online softmax: each row lives in one 16-lane half of one C register
#pragma unroll
    for (int r = 0; r < 8; ++r) {
      float mx = fmaxf(S0[r], S1[r]);
      mx = fmaxf(mx, __shfl_xor(mx, 8, 32));
      mx = fmaxf(mx, __shfl_xor(mx, 4, 32));
      mx = fmaxf(mx, __shfl_xor(mx, 2, 32));
      mx = fmaxf(mx, __shfl_xor(mx, 1, 32));
      float mn = fmaxf(mrow[r], mx);
      float p0 = __expf(S0[r] - mn);
      float p1 = __expf(S1[r] - mn);
      float ps = p0 + p1;
      ps += __shfl_xor(ps, 8, 32);
      ps += __shfl_xor(ps, 4, 32);
      ps += __shfl_xor(ps, 2, 32);
      ps += __shfl_xor(ps, 1, 32);
      float alpha = __expf(mrow[r] - mn);
      lrow[r] = lrow[r] * alpha + ps;
      mrow[r] = mn;
#pragma unroll
      for (int nt = 0; nt < 4; ++nt) O[nt][r] *= alpha;
      int prow = r + hf * 8;
      P[prow * 32 + ln]      = (_Float16)p0;
      P[prow * 32 + 16 + ln] = (_Float16)p1;
    }

    asm volatile("s_wait_dscnt 0" ::: "memory");

    // P (16x32) back as A operand
    const _Float16* pr = P + ln * 32 + hf * 16;
    v16h aP = mk16(*(const h8*)pr, *(const h8*)(pr + 8));

    // O += P @ V (V tiles already resident)
#pragma unroll
    for (int nt = 0; nt < 4; ++nt) {
      O[nt] = __builtin_amdgcn_wmma_f32_16x16x32_f16(
          false, aP, false, bV[nt], (short)0, O[nt], false, false);
    }
  }

  float* op = out + (size_t)(b * Tt + i0) * Hh;
#pragma unroll
  for (int r = 0; r < 8; ++r) {
    float inv = 1.f / lrow[r];
    int row = r + hf * 8;
#pragma unroll
    for (int nt = 0; nt < 4; ++nt)
      op[row * Hh + nt * 16 + ln] = O[nt][r] * inv;
  }
}

// ---------------------------------------------------------------------------
extern "C" void kernel_launch(void* const* d_in, const int* in_sizes, int n_in,
                              void* d_out, int out_size, void* d_ws, size_t ws_size,
                              hipStream_t stream) {
  const float* x  = (const float*)d_in[0];
  const float* Wk = (const float*)d_in[1];
  const float* bk = (const float*)d_in[2];
  const float* Wq = (const float*)d_in[3];
  const float* bq = (const float*)d_in[4];
  const float* Wv = (const float*)d_in[5];
  const float* bv = (const float*)d_in[6];

  const size_t NT = (size_t)Bb * Tt * Hh;  // 1,048,576 halfs per tensor
  _Float16* ws  = (_Float16*)d_ws;
  _Float16* qf  = ws;
  _Float16* kf  = ws + NT;
  _Float16* vt  = ws + 2 * NT;
  _Float16* WqT = ws + 3 * NT;
  _Float16* WkT = WqT + Hh * Dd;
  _Float16* WvT = WkT + Hh * Dd;

  wprep_k<<<(Hh * Dd + 255) / 256, 256, 0, stream>>>(Wk, Wq, Wv, WkT, WqT, WvT);

  // B*T/16 = 1024 row tiles, 8 waves per 256-thread block
  qkvproj_k<<<128, 256, 0, stream>>>(x, WqT, WkT, WvT, bq, bk, bv, qf, kf, vt);

  attn_k<<<128, 256, 0, stream>>>(qf, kf, vt, (float*)d_out);
}
